// BinaryResNetBlock_72206990181172
// MI455X (gfx1250) — 8-bit matrix operand order repaired and checked by reading, compiled, not run on hardware
//
#include <hip/hip_runtime.h>

typedef int v8i __attribute__((ext_vector_type(8)));
typedef unsigned long long u64ll;

#define N_IMG   64
#define HW_     3136
#define W_      56
#define PIX     (N_IMG * HW_)
#define KTOT    2304
#define NSTEPS  36
#define EPS_BN  1e-5

__global__ __launch_bounds__(256) void k_zero(u64ll* p){
  p[blockIdx.x * 256 + threadIdx.x] = 0ull;
}

__global__ __launch_bounds__(256) void k_packw(const float* __restrict__ w,
                                               signed char* __restrict__ wt,
                                               float* __restrict__ scale){
  __shared__ float red[256];
  const int o = blockIdx.x, t = threadIdx.x;
  float acc = 0.f;
  #pragma unroll
  for (int it = 0; it < 9; ++it){
    const int i = t + it * 256;
    const int khw = i >> 8, c = i & 255;
    const float v = w[o * KTOT + c * 9 + khw];
    acc += fabsf(v);
    wt[o * KTOT + i] = (signed char)((v > 0.f) - (v < 0.f));
  }
  red[t] = acc;
  __syncthreads();
  for (int s = 128; s > 0; s >>= 1){
    if (t < s) red[t] += red[t + s];
    __syncthreads();
  }
  if (t == 0) scale[o] = red[0] * (1.f / (float)KTOT);
}

__global__ __launch_bounds__(256) void k_binarize(const float* __restrict__ x,
                                                  signed char* __restrict__ sx){
  const long long idx = (long long)blockIdx.x * 256 + threadIdx.x;
  const int o  = (int)(idx & 255);
  const int p  = (int)(idx >> 8);
  const int n  = p / HW_;
  const int hw = p - n * HW_;
  const float v = x[(size_t)(n * 256 + o) * HW_ + hw];
  sx[idx] = (signed char)((v > 0.f) - (v < 0.f));
}

__global__ __launch_bounds__(256) void k_bgemm(const signed char* __restrict__ act,
                                               const signed char* __restrict__ wt,
                                               short* __restrict__ z,
                                               u64ll* __restrict__ cS,
                                               u64ll* __restrict__ cQ){
  __shared__ signed char As[128 * 72];
  __shared__ signed char Bs[128 * 72];
  __shared__ u64ll rS[128];
  __shared__ u64ll rQ[128];

  const int t     = threadIdx.x;
  const int pBase = blockIdx.x * 128;
  const int oBase = blockIdx.y * 128;
  if (t < 128){ rS[t] = 0ull; rQ[t] = 0ull; }

  const int r0  = t >> 2;
  const int r1  = r0 + 64;
  const int c16 = (t & 3) * 16;
  const int p0 = pBase + r0, p1 = pBase + r1;
  const int n0 = p0 / HW_, hr0 = p0 - n0 * HW_;
  const int n1 = p1 / HW_, hr1 = p1 - n1 * HW_;
  const int h0 = hr0 / W_, x0 = hr0 - h0 * W_;
  const int h1 = hr1 / W_, x1 = hr1 - h1 * W_;
  const signed char* aImg0 = act + (size_t)n0 * (HW_ * 256);
  const signed char* aImg1 = act + (size_t)n1 * (HW_ * 256);
  const signed char* bRow0 = wt + (size_t)(oBase + r0) * KTOT + c16;
  const signed char* bRow1 = wt + (size_t)(oBase + r1) * KTOT + c16;

  const int lane = t & 31;
  const int l15  = lane & 15, lhi = lane >> 4;
  const int wv   = t >> 5;
  const int wm   = (wv & 3) * 32;
  const int wn   = (wv >> 2) * 64;

  const v8i vzero = {0,0,0,0,0,0,0,0};
  v8i acc[2][4];
  #pragma unroll
  for (int i = 0; i < 2; ++i)
    #pragma unroll
    for (int j = 0; j < 4; ++j) acc[i][j] = vzero;

  for (int kk = 0; kk < NSTEPS; ++kk){
    const int khw = kk >> 2;
    const int c0  = (kk & 3) * 64;
    const int dh  = khw / 3 - 1;
    const int dw  = khw - (khw / 3) * 3 - 1;

    __syncthreads();

    {
      const int hh = h0 + dh, ww = x0 + dw;
      int4 v = make_int4(0, 0, 0, 0);
      if ((unsigned)hh < (unsigned)W_ && (unsigned)ww < (unsigned)W_)
        v = *(const int4*)(aImg0 + (hh * W_ + ww) * 256 + c0 + c16);
      *(int2*)(As + r0 * 72 + c16)     = make_int2(v.x, v.y);
      *(int2*)(As + r0 * 72 + c16 + 8) = make_int2(v.z, v.w);
    }
    {
      const int hh = h1 + dh, ww = x1 + dw;
      int4 v = make_int4(0, 0, 0, 0);
      if ((unsigned)hh < (unsigned)W_ && (unsigned)ww < (unsigned)W_)
        v = *(const int4*)(aImg1 + (hh * W_ + ww) * 256 + c0 + c16);
      *(int2*)(As + r1 * 72 + c16)     = make_int2(v.x, v.y);
      *(int2*)(As + r1 * 72 + c16 + 8) = make_int2(v.z, v.w);
    }
    {
      const int4 v = *(const int4*)(bRow0 + kk * 64);
      *(int2*)(Bs + r0 * 72 + c16)     = make_int2(v.x, v.y);
      *(int2*)(Bs + r0 * 72 + c16 + 8) = make_int2(v.z, v.w);
      const int4 u = *(const int4*)(bRow1 + kk * 64);
      *(int2*)(Bs + r1 * 72 + c16)     = make_int2(u.x, u.y);
      *(int2*)(Bs + r1 * 72 + c16 + 8) = make_int2(u.z, u.w);
    }
    if (kk + 1 < NSTEPS){
      __builtin_prefetch(bRow0 + (kk + 1) * 64, 0, 1);
      __builtin_prefetch(bRow1 + (kk + 1) * 64, 0, 1);
    }
    __syncthreads();

    v8i af[2], bf[4];
    #pragma unroll
    for (int i = 0; i < 2; ++i){
      const int* ip = (const int*)(As + (wm + i * 16 + l15) * 72 + lhi * 8);
      v8i f;
      f[0] = ip[0]; f[1] = ip[1]; f[2] = ip[4];  f[3] = ip[5];
      f[4] = ip[8]; f[5] = ip[9]; f[6] = ip[12]; f[7] = ip[13];
      af[i] = f;
    }
    #pragma unroll
    for (int j = 0; j < 4; ++j){
      const int* ip = (const int*)(Bs + (wn + j * 16 + l15) * 72 + lhi * 8);
      v8i f;
      f[0] = ip[0]; f[1] = ip[1]; f[2] = ip[4];  f[3] = ip[5];
      f[4] = ip[8]; f[5] = ip[9]; f[6] = ip[12]; f[7] = ip[13];
      bf[j] = f;
    }
    #pragma unroll
    for (int i = 0; i < 2; ++i)
      #pragma unroll
      for (int j = 0; j < 4; ++j)
        acc[i][j] = __builtin_amdgcn_wmma_i32_16x16x64_iu8(
            true, af[i], true, bf[j], acc[i][j], false, false);
  }

  #pragma unroll
  for (int j = 0; j < 4; ++j){
    const int oc = oBase + wn + j * 16 + l15;
    u64ll s = 0ull, q = 0ull;
    #pragma unroll
    for (int i = 0; i < 2; ++i){
      const int pr = pBase + wm + i * 16 + lhi * 8;
      #pragma unroll
      for (int g = 0; g < 8; ++g){
        const int v = acc[i][j][g];
        z[(pr + g) * 256 + oc] = (short)v;
        s += (u64ll)(long long)v;
        q += (u64ll)((long long)v * (long long)v);
      }
    }
    atomicAdd(&rS[wn + j * 16 + l15], s);
    atomicAdd(&rQ[wn + j * 16 + l15], q);
  }
  __syncthreads();
  if (t < 128){
    atomicAdd(&cS[oBase + t], rS[t]);
    atomicAdd(&cQ[oBase + t], rQ[t]);
  }
}

__global__ __launch_bounds__(256) void k_bncoef(const u64ll* __restrict__ cS,
                                                const u64ll* __restrict__ cQ,
                                                const float* __restrict__ scale,
                                                const float* __restrict__ gamma,
                                                const float* __restrict__ beta,
                                                float* __restrict__ ab){
  const int o = threadIdx.x;
  const double cnt = (double)PIX;
  const double mz  = (double)(long long)cS[o] / cnt;
  const double mq  = (double)(long long)cQ[o] / cnt;
  double varz = mq - mz * mz;
  if (varz < 0.0) varz = 0.0;
  const double s    = (double)scale[o];
  const double istd = 1.0 / sqrt(s * s * varz + EPS_BN);
  const double ga   = (double)gamma[o];
  ab[o]       = (float)(ga * s * istd);
  ab[o + 256] = (float)((double)beta[o] - ga * s * mz * istd);
}

__global__ __launch_bounds__(256) void k_bnsign(const short* __restrict__ z,
                                                const float* __restrict__ ab,
                                                signed char* __restrict__ sy){
  const long long base = (long long)blockIdx.x * 1024 + threadIdx.x;
  #pragma unroll
  for (int u = 0; u < 4; ++u){
    const long long idx = base + (long long)u * 256;
    const int o = (int)(idx & 255);
    const float v = ab[o] * (float)z[idx] + ab[o + 256];
    sy[idx] = (signed char)((v > 0.f) - (v < 0.f));
  }
}

__global__ __launch_bounds__(256) void k_final(const short* __restrict__ z,
                                               const float* __restrict__ ab,
                                               const float* __restrict__ x,
                                               float* __restrict__ out){
  const long long base = (long long)blockIdx.x * 1024 + threadIdx.x;
  #pragma unroll
  for (int u = 0; u < 4; ++u){
    const long long idx = base + (long long)u * 256;
    const int hw = (int)(idx % HW_);
    const int no = (int)(idx / HW_);
    const int o  = no & 255;
    const int n  = no >> 8;
    const int p  = n * HW_ + hw;
    const float v = ab[o] * (float)z[(size_t)p * 256 + o] + ab[o + 256] + x[idx];
    out[idx] = (float)((v > 0.f) - (v < 0.f));
  }
}

extern "C" void kernel_launch(void* const* d_in, const int* in_sizes, int n_in,
                              void* d_out, int out_size, void* d_ws, size_t ws_size,
                              hipStream_t stream){
  (void)in_sizes; (void)n_in; (void)out_size; (void)ws_size;
  const float* x  = (const float*)d_in[0];
  const float* w1 = (const float*)d_in[1];
  const float* g1 = (const float*)d_in[2];
  const float* b1 = (const float*)d_in[3];
  const float* w2 = (const float*)d_in[4];
  const float* g2 = (const float*)d_in[5];
  const float* b2 = (const float*)d_in[6];
  float* out = (float*)d_out;
  char* ws = (char*)d_ws;

  size_t off = 0;
  auto carve = [&](size_t bytes) -> char* {
    char* p = ws + off;
    off += (bytes + 255) & ~(size_t)255;
    return p;
  };
  signed char* act = (signed char*)carve((size_t)PIX * 256);
  short*       z   = (short*)      carve((size_t)PIX * 256 * 2);
  signed char* wt1 = (signed char*)carve((size_t)256 * KTOT);
  signed char* wt2 = (signed char*)carve((size_t)256 * KTOT);
  float* s1 = (float*)carve(256 * 4);
  float* s2 = (float*)carve(256 * 4);
  u64ll* sums = (u64ll*)carve(4 * 256 * 8);
  u64ll* cS1 = sums, *cQ1 = sums + 256, *cS2 = sums + 512, *cQ2 = sums + 768;
  float* ab1 = (float*)carve(512 * 4);
  float* ab2 = (float*)carve(512 * 4);

  k_zero    <<<4, 256, 0, stream>>>(sums);
  k_packw   <<<256, 256, 0, stream>>>(w1, wt1, s1);
  k_packw   <<<256, 256, 0, stream>>>(w2, wt2, s2);
  k_binarize<<<PIX, 256, 0, stream>>>(x, act);

  k_bgemm   <<<dim3(PIX / 128, 2), 256, 0, stream>>>(act, wt1, z, cS1, cQ1);
  k_bncoef  <<<1, 256, 0, stream>>>(cS1, cQ1, s1, g1, b1, ab1);
  k_bnsign  <<<PIX * 256 / 1024, 256, 0, stream>>>(z, ab1, act);

  k_bgemm   <<<dim3(PIX / 128, 2), 256, 0, stream>>>(act, wt2, z, cS2, cQ2);
  k_bncoef  <<<1, 256, 0, stream>>>(cS2, cQ2, s2, g2, b2, ab2);
  k_final   <<<PIX * 256 / 1024, 256, 0, stream>>>(z, ab2, x, out);
}
